// SignalProp_8495445311659
// MI455X (gfx1250) — matrix operand order repaired and checked by reading, not compiled, not run on hardware
//
#include <hip/hip_runtime.h>
#include <hip/hip_bf16.h>
#include <math.h>

typedef _Float16 half_t;
typedef __attribute__((ext_vector_type(16))) _Float16 v16h;
typedef __attribute__((ext_vector_type(8)))  _Float16 v8h;
typedef __attribute__((ext_vector_type(8)))  float    v8f;

#define N_NODES 150000
#define E_NETC  400000
#define E_CELLC 200000
#define XS 72            // activation tile LDS stride in halves (144B, 16B-aligned rows)

// ---------------- workspace layout (floats; all 12-wide arrays 16B aligned) ----------------
#define OFF_NETSUM 0ull                     // 150000*12
#define OFF_NETDEG 1800000ull               // 150000
#define OFF_CNT    1950000ull               // 150000
#define OFF_MSUM   2100000ull               // 150000*12
#define OFF_KVS    3900000ull               // 144
#define OFF_KSSUM  3900144ull               // 12
#define OFF_SUMSQ  3900156ull               // 2  (q, k)
#define ZERO_COUNT 3900158ull
#define OFF_MMAX   3900160ull               // 150000*12
#define OFF_KVSS   5700160ull               // 144
#define OFF_KSS    5700304ull               // 12
#define OFF_SCAL   5700316ull               // 4
#define OFF_Q      5700320ull               // 200000*12
#define OFF_QJI    8100320ull               // 200000*12
#define OFF_V      10500320ull              // 200000*12

struct MlpPtrs { const float *W0,*b0,*W1,*b1,*W2,*b2,*W3,*b3,*W4,*b4; };

// ---------------- WMMA helpers ----------------
__device__ static inline v8f wmma16(v16h a, v16h b, v8f c) {
  return __builtin_amdgcn_wmma_f32_16x16x32_f16(false, a, false, b, (short)0, c, false, false);
}

// A-fragment (16x32 f16) from LDS row-major tile. Contiguous K (matches B):
// g=lane/16, row=lane%16: a[0..15]=K[kbase+g*16 .. +15]
__device__ static inline v16h load_a_frag(const half_t* xt, int stride, int kbase) {
  const int lane = threadIdx.x & 31;
  const int g = lane >> 4;
  const int row = lane & 15;
  const half_t* p = xt + row * stride + kbase + g * 16;
  v8h lo = *(const v8h*)p;
  v8h hi = *(const v8h*)(p + 8);
  v16h a;
#pragma unroll
  for (int i = 0; i < 8; ++i) { a[i] = lo[i]; a[i + 8] = hi[i]; }
  return a;
}

// B-fragment (32x16 f16) from LDS weights stored TRANSPOSED [N][Kpad]:
// lane holds col=lane%16, K=[kbase + g*16 .. +15]  -> one contiguous 32B read
__device__ static inline v16h load_b_frag(const half_t* Wt, int kstride, int ncol0, int kbase) {
  const int lane = threadIdx.x & 31;
  const int g = lane >> 4;
  const int col = lane & 15;
  return *(const v16h*)(Wt + (size_t)(ncol0 + col) * kstride + kbase + g * 16);
}

__device__ static inline v8f bias_frag(const float* B16) {
  const float bv = B16[threadIdx.x & 15];
  v8f c;
#pragma unroll
  for (int r = 0; r < 8; ++r) c[r] = bv;
  return c;
}

// ---------------- vectorized gather/convert helpers ----------------
__device__ static inline void st4h(half_t* p, float4 x) {
  p[0] = (half_t)x.x; p[1] = (half_t)x.y; p[2] = (half_t)x.z; p[3] = (half_t)x.w;
}
// nf row (10 floats, 8B aligned) -> f16
__device__ static inline void cvt_nf_row(half_t* dst, const float* src) {
  const float2* s = (const float2*)src;
#pragma unroll
  for (int j = 0; j < 5; ++j) { float2 x = s[j]; dst[2*j] = (half_t)x.x; dst[2*j+1] = (half_t)x.y; }
}
// gt row (12 = ats|slew|netd, each 16B aligned) -> f16
__device__ static inline void cvt_gt_row(half_t* dst, const float* A, const float* S,
                                         const float* D, int n) {
  st4h(dst,     *(const float4*)(A + n * 4));
  st4h(dst + 4, *(const float4*)(S + n * 4));
  st4h(dst + 8, *(const float4*)(D + n * 4));
}
// 12 floats from 16B-aligned row
__device__ static inline void ld12(float* o, const float* p) {
  float4 t0 = ((const float4*)p)[0], t1 = ((const float4*)p)[1], t2 = ((const float4*)p)[2];
  o[0]=t0.x; o[1]=t0.y; o[2]=t0.z; o[3]=t0.w;
  o[4]=t1.x; o[5]=t1.y; o[6]=t1.z; o[7]=t1.w;
  o[8]=t2.x; o[9]=t2.y; o[10]=t2.z; o[11]=t2.w;
}

// convert f32 weights [Ksrc][Nsrc] -> LDS f16 transposed [Npad][Kpad], zero padded
__device__ static inline void load_wt(half_t* dst, const float* W, int Ksrc, int Nsrc,
                                      int Kpad, int Npad) {
  for (int i = threadIdx.x; i < Npad * Kpad; i += blockDim.x) {
    int n = i / Kpad, k = i - n * Kpad;
    float v = (k < Ksrc && n < Nsrc) ? W[k * Nsrc + n] : 0.f;
    dst[i] = (half_t)v;
  }
}
__device__ static inline void load_bias(float* dst, const float* b, int nsrc, int npad) {
  for (int i = threadIdx.x; i < npad; i += blockDim.x)
    dst[i] = (i < nsrc) ? b[i] : 0.f;
}

// 5-layer MLP on a 16x64 LDS tile: 4x (64->64, leaky relu 0.2) + (64->16 final, linear).
__device__ static inline v8f mlp5_64(half_t* xt, const half_t* Wt4, const float* Bs,
                                     const half_t* W4t, const float* B4) {
  const int lane = threadIdx.x & 31;
  const int g = lane >> 4;
  const int col = lane & 15;
  for (int l = 0; l < 4; ++l) {
    const half_t* W = Wt4 + l * 64 * 64;
    const float* B = Bs + l * 64;
    v16h a0 = load_a_frag(xt, XS, 0);
    v16h a1 = load_a_frag(xt, XS, 32);
    v8f c0 = bias_frag(B + 0);
    v8f c1 = bias_frag(B + 16);
    v8f c2 = bias_frag(B + 32);
    v8f c3 = bias_frag(B + 48);
    c0 = wmma16(a0, load_b_frag(W, 64, 0, 0), c0);
    c0 = wmma16(a1, load_b_frag(W, 64, 0, 32), c0);
    c1 = wmma16(a0, load_b_frag(W, 64, 16, 0), c1);
    c1 = wmma16(a1, load_b_frag(W, 64, 16, 32), c1);
    c2 = wmma16(a0, load_b_frag(W, 64, 32, 0), c2);
    c2 = wmma16(a1, load_b_frag(W, 64, 32, 32), c2);
    c3 = wmma16(a0, load_b_frag(W, 64, 48, 0), c3);
    c3 = wmma16(a1, load_b_frag(W, 64, 48, 32), c3);
#pragma unroll
    for (int r = 0; r < 8; ++r) {
      half_t* xr = xt + (r + 8 * g) * XS;
      float y0 = c0[r]; y0 = y0 > 0.f ? y0 : 0.2f * y0;
      float y1 = c1[r]; y1 = y1 > 0.f ? y1 : 0.2f * y1;
      float y2 = c2[r]; y2 = y2 > 0.f ? y2 : 0.2f * y2;
      float y3 = c3[r]; y3 = y3 > 0.f ? y3 : 0.2f * y3;
      xr[col] = (half_t)y0; xr[16 + col] = (half_t)y1;
      xr[32 + col] = (half_t)y2; xr[48 + col] = (half_t)y3;
    }
  }
  v16h a0 = load_a_frag(xt, XS, 0);
  v16h a1 = load_a_frag(xt, XS, 32);
  v8f c = bias_frag(B4);
  c = wmma16(a0, load_b_frag(W4t, 64, 0, 0), c);
  c = wmma16(a1, load_b_frag(W4t, 64, 0, 32), c);
  return c;
}

__device__ static inline float gt_at(const float* A, const float* S, const float* D, int n, int c) {
  return (c < 4) ? A[n * 4 + c] : (c < 8) ? S[n * 4 + c - 4] : D[n * 4 + c - 8];
}

__device__ static inline void atomicMaxFloat(float* addr, float v) {
  if (v >= 0.f) atomicMax((int*)addr, __float_as_int(v));
  else          atomicMin((unsigned int*)addr, __float_as_uint(v));
}

// ---------------- kernels ----------------
__global__ void init_ws_kernel(float* ws) {
  size_t i = (size_t)blockIdx.x * blockDim.x + threadIdx.x;
  size_t stride = (size_t)gridDim.x * blockDim.x;
  for (size_t j = i; j < ZERO_COUNT; j += stride) ws[j] = 0.f;
  for (size_t j = i; j < 1800000ull; j += stride) ws[OFF_MMAX + j] = -3.402823466e38f;
}

__global__ void degrees_kernel(const int* net_dst, float* net_deg,
                               const int* cell_dst, float* cnt) {
  int i = blockIdx.x * blockDim.x + threadIdx.x;
  if (i < E_NETC)  atomicAdd(&net_deg[net_dst[i]], 1.f);
  if (i < E_CELLC) atomicAdd(&cnt[cell_dst[i]], 1.f);
}

// netprop on net edges -> atomic segment sum into net_sum
__global__ __launch_bounds__(256) void netprop_kernel(
    const float* nf, const float* n_ats, const float* n_slew, const float* n_netd,
    const int* net_src, const int* net_dst, MlpPtrs mp, float* net_sum, int ntiles) {
  __shared__ __align__(32) half_t Wt[4 * 64 * 64];
  __shared__ __align__(32) half_t W4t[16 * 64];
  __shared__ float Bs[4 * 64];
  __shared__ float B4[16];
  __shared__ __align__(32) half_t Xt[8][16 * XS];
  __shared__ int Didx[8][16];
  load_wt(Wt + 0 * 4096, mp.W0, 32, 64, 64, 64);
  load_wt(Wt + 1 * 4096, mp.W1, 64, 64, 64, 64);
  load_wt(Wt + 2 * 4096, mp.W2, 64, 64, 64, 64);
  load_wt(Wt + 3 * 4096, mp.W3, 64, 64, 64, 64);
  load_wt(W4t, mp.W4, 64, 12, 64, 16);
  load_bias(Bs + 0, mp.b0, 64, 64); load_bias(Bs + 64, mp.b1, 64, 64);
  load_bias(Bs + 128, mp.b2, 64, 64); load_bias(Bs + 192, mp.b3, 64, 64);
  load_bias(B4, mp.b4, 12, 16);
  __syncthreads();
  const int wv = threadIdx.x >> 5;
  const int tile = blockIdx.x * 8 + wv;
  if (tile >= ntiles) return;
  const int e0 = tile * 16;
  const int lane = threadIdx.x & 31;
  half_t* xt = Xt[wv];
  if (lane < 16) {
    int e = e0 + lane;
    int s = net_src[e], d = net_dst[e];
    Didx[wv][lane] = d;
    half_t* xr = xt + lane * XS;
    cvt_gt_row(xr, n_ats, n_slew, n_netd, s);
    cvt_nf_row(xr + 12, nf + s * 10);
    cvt_nf_row(xr + 22, nf + d * 10);
    for (int j = 32; j < 64; ++j) xr[j] = (half_t)0.f;
  }
  v8f acc = mlp5_64(xt, Wt, Bs, W4t, B4);
  const int g = lane >> 4, col = lane & 15;
  if (col < 12) {
#pragma unroll
    for (int r = 0; r < 8; ++r) {
      int d = Didx[wv][r + 8 * g];
      atomicAdd(&net_sum[d * 12 + col], acc[r]);
    }
  }
}

// mlp_q -> Qji, q = Qji@Wq+bq, accumulate sum(q^2)
__global__ __launch_bounds__(256) void mlpq_kernel(
    const float* nf, const float* n_ats, const float* n_slew, const float* n_netd,
    const int* cell_src, const int* cell_dst, MlpPtrs mp,
    const float* Wq, const float* bq,
    float* q_out, float* Qji_out, float* sumsq_q, int ntiles) {
  __shared__ __align__(32) half_t Wt[4 * 64 * 64];
  __shared__ __align__(32) half_t W4t[16 * 64];
  __shared__ __align__(32) half_t Wqt[16 * 32];
  __shared__ float Bs[4 * 64];
  __shared__ float B4[16];
  __shared__ float Bq[16];
  __shared__ __align__(32) half_t Xt[8][16 * XS];
  __shared__ int Sidx[8][16];
  load_wt(Wt + 0 * 4096, mp.W0, 32, 64, 64, 64);
  load_wt(Wt + 1 * 4096, mp.W1, 64, 64, 64, 64);
  load_wt(Wt + 2 * 4096, mp.W2, 64, 64, 64, 64);
  load_wt(Wt + 3 * 4096, mp.W3, 64, 64, 64, 64);
  load_wt(W4t, mp.W4, 64, 12, 64, 16);
  load_wt(Wqt, Wq, 12, 12, 32, 16);
  load_bias(Bs + 0, mp.b0, 64, 64); load_bias(Bs + 64, mp.b1, 64, 64);
  load_bias(Bs + 128, mp.b2, 64, 64); load_bias(Bs + 192, mp.b3, 64, 64);
  load_bias(B4, mp.b4, 12, 16); load_bias(Bq, bq, 12, 16);
  __syncthreads();
  const int wv = threadIdx.x >> 5;
  const int tile = blockIdx.x * 8 + wv;
  if (tile >= ntiles) return;
  const int e0 = tile * 16;
  const int lane = threadIdx.x & 31;
  half_t* xt = Xt[wv];
  if (lane < 16) {
    int e = e0 + lane;
    int s = cell_src[e];
    Sidx[wv][lane] = s;
    half_t* xr = xt + lane * XS;
    cvt_nf_row(xr, nf + s * 10);
    cvt_nf_row(xr + 10, nf + cell_dst[e] * 10);
    cvt_gt_row(xr + 20, n_ats, n_slew, n_netd, s);
    for (int j = 32; j < 64; ++j) xr[j] = (half_t)0.f;
  }
  v8f acc = mlp5_64(xt, Wt, Bs, W4t, B4);
  const int g = lane >> 4, col = lane & 15;
  // Qji = acc + last_nf ; stash into xt[:,0:32] (zero padded) for q projection
#pragma unroll
  for (int r = 0; r < 8; ++r) {
    int row = r + 8 * g;
    float qv = 0.f;
    if (col < 12) {
      int s = Sidx[wv][row];
      qv = acc[r] + gt_at(n_ats, n_slew, n_netd, s, col);
      Qji_out[(size_t)(e0 + row) * 12 + col] = qv;
    }
    xt[row * XS + col] = (half_t)qv;
    xt[row * XS + 16 + col] = (half_t)0.f;
  }
  v8f qc = bias_frag(Bq);
  qc = wmma16(load_a_frag(xt, XS, 0), load_b_frag(Wqt, 32, 0, 0), qc);
  float ss = 0.f;
#pragma unroll
  for (int r = 0; r < 8; ++r) {
    if (col < 12) {
      q_out[(size_t)(e0 + r + 8 * g) * 12 + col] = qc[r];
      ss += qc[r] * qc[r];
    }
  }
  for (int off = 16; off > 0; off >>= 1) ss += __shfl_xor(ss, off, 32);
  if (lane == 0) atomicAdd(sumsq_q, ss);
}

// k = ef[:,:120]@Wk+bk, v = ef[:,120:]@Wv+bv; accumulate sum(k^2), sum_k, k^T v; store v
__global__ __launch_bounds__(256) void kv_kernel(
    const float* ef, const float* Wk, const float* bk, const float* Wv, const float* bvp,
    float* v_out, float* sumsq_k, float* ks_sum, float* kvs, int ntiles) {
  __shared__ __align__(32) half_t Wkt[16 * 128];
  __shared__ __align__(32) half_t Wvt[16 * 416];
  __shared__ float Bk[16], Bv[16];
  __shared__ __align__(32) half_t Ct[8][16 * 40];
  __shared__ float Ktile[8][256];
  __shared__ float Vtile[8][256];
  load_wt(Wkt, Wk, 120, 12, 128, 16);
  load_wt(Wvt, Wv, 392, 12, 416, 16);
  load_bias(Bk, bk, 12, 16); load_bias(Bv, bvp, 12, 16);
  __syncthreads();
  const int wv = threadIdx.x >> 5;
  const int tile = blockIdx.x * 8 + wv;
  if (tile >= ntiles) return;
  const int e0 = tile * 16;
  const int lane = threadIdx.x & 31;
  half_t* ct = Ct[wv];
  // warm caches with this tile's contiguous 16x512-f32 ef block (32KB):
  // 8 prefetch ops x 32 lanes spaced 128B (global_prefetch_b8)
#pragma unroll
  for (int pb = 0; pb < 8; ++pb) {
    const float* pp = ef + (size_t)(e0 + (lane >> 1)) * 512 + (lane & 1) * 256 + pb * 32;
    __builtin_prefetch(pp, 0, 3);
  }
  v8f kc = bias_frag(Bk);
  for (int kt = 0; kt < 4; ++kt) {                 // 120 = 3*32 + 24 (pad to 128)
    for (int i = lane; i < 128; i += 32) {         // 16 rows x 8 float4
      int r = i >> 3, c4 = (i & 7) << 2;
      int kcol = kt * 32 + c4;
      float4 x = make_float4(0.f, 0.f, 0.f, 0.f);
      if (kcol < 120) x = *(const float4*)(ef + (size_t)(e0 + r) * 512 + kcol);
      st4h(ct + r * 40 + c4, x);
    }
    kc = wmma16(load_a_frag(ct, 40, 0), load_b_frag(Wkt, 128, 0, kt * 32), kc);
  }
  v8f vc = bias_frag(Bv);
  for (int kt = 0; kt < 13; ++kt) {                // 392 = 12*32 + 8 (pad to 416)
    for (int i = lane; i < 128; i += 32) {
      int r = i >> 3, c4 = (i & 7) << 2;
      int kcol = kt * 32 + c4;
      float4 x = make_float4(0.f, 0.f, 0.f, 0.f);
      if (kcol < 392) x = *(const float4*)(ef + (size_t)(e0 + r) * 512 + 120 + kcol);
      st4h(ct + r * 40 + c4, x);
    }
    vc = wmma16(load_a_frag(ct, 40, 0), load_b_frag(Wvt, 416, 0, kt * 32), vc);
  }
  const int g = lane >> 4, col = lane & 15;
  float ssk = 0.f, ksp = 0.f;
#pragma unroll
  for (int r = 0; r < 8; ++r) {
    int row = r + 8 * g;
    float kvv = (col < 12) ? kc[r] : 0.f;
    float vvv = (col < 12) ? vc[r] : 0.f;
    Ktile[wv][row * 16 + col] = kvv;
    Vtile[wv][row * 16 + col] = vvv;
    if (col < 12) {
      v_out[(size_t)(e0 + row) * 12 + col] = vc[r];
      ssk += kvv * kvv; ksp += kvv;
    }
  }
  for (int off = 16; off > 0; off >>= 1) ssk += __shfl_xor(ssk, off, 32);
  if (lane == 0) atomicAdd(sumsq_k, ssk);
  ksp += __shfl_xor(ksp, 16, 32);
  if (lane < 12) atomicAdd(&ks_sum[lane], ksp);
  for (int idx = lane; idx < 144; idx += 32) {
    int m = idx / 12, nn = idx - m * 12;
    float s = 0.f;
    for (int r = 0; r < 16; ++r) s += Ktile[wv][r * 16 + m] * Vtile[wv][r * 16 + nn];
    atomicAdd(&kvs[idx], s);
  }
}

__global__ void finalize_attn(const float* sums, const float* kvs, const float* ks,
                              float* kvs_s, float* ks_s, float* scal) {
  __shared__ float inv;
  if (threadIdx.x == 0) {
    inv = 1.f / (sqrtf(sums[0]) * sqrtf(sums[1]));
    scal[0] = (float)E_CELLC;
  }
  __syncthreads();
  if (threadIdx.x < 144) kvs_s[threadIdx.x] = kvs[threadIdx.x] * inv;
  if (threadIdx.x < 12)  ks_s[threadIdx.x] = ks[threadIdx.x] * inv;
}

// attention + layernorm + mlp_m -> atomic msum/mmax, efce out
__global__ __launch_bounds__(256) void mlpm_kernel(
    const float* nf, const float* n_ats, const float* n_slew, const float* n_netd,
    const int* cell_src, const int* cell_dst,
    const float* q_in, const float* Qji_in, const float* v_in,
    const float* kvs_s, const float* ks_s, const float* scal,
    const float* ln_g, const float* ln_b, MlpPtrs mp,
    float* msum, float* mmax, float* efce_out, int ntiles) {
  __shared__ __align__(32) half_t Wt[4 * 64 * 64];
  __shared__ __align__(32) half_t W4t[16 * 64];
  __shared__ float Bs[4 * 64];
  __shared__ float B4[16];
  __shared__ __align__(32) half_t Xt[8][16 * XS];
  __shared__ int Sidx[8][16], Didx[8][16];
  __shared__ float Kvs[144], Kss[12], Lg[12], Lb[12], Lf;
  load_wt(Wt + 0 * 4096, mp.W0, 44, 64, 64, 64);
  load_wt(Wt + 1 * 4096, mp.W1, 64, 64, 64, 64);
  load_wt(Wt + 2 * 4096, mp.W2, 64, 64, 64, 64);
  load_wt(Wt + 3 * 4096, mp.W3, 64, 64, 64, 64);
  load_wt(W4t, mp.W4, 64, 16, 64, 16);
  load_bias(Bs + 0, mp.b0, 64, 64); load_bias(Bs + 64, mp.b1, 64, 64);
  load_bias(Bs + 128, mp.b2, 64, 64); load_bias(Bs + 192, mp.b3, 64, 64);
  load_bias(B4, mp.b4, 16, 16);
  if (threadIdx.x < 144) Kvs[threadIdx.x] = kvs_s[threadIdx.x];
  if (threadIdx.x < 12) { Kss[threadIdx.x] = ks_s[threadIdx.x]; Lg[threadIdx.x] = ln_g[threadIdx.x]; Lb[threadIdx.x] = ln_b[threadIdx.x]; }
  if (threadIdx.x == 0) Lf = scal[0];
  __syncthreads();
  const int wv = threadIdx.x >> 5;
  const int tile = blockIdx.x * 8 + wv;
  if (tile >= ntiles) return;
  const int e0 = tile * 16;
  const int lane = threadIdx.x & 31;
  half_t* xt = Xt[wv];
  if (lane < 16) {
    int e = e0 + lane;
    int s = cell_src[e], d = cell_dst[e];
    Sidx[wv][lane] = s; Didx[wv][lane] = d;
    float q[12], Q[12], v[12];
    ld12(q, q_in + (size_t)e * 12);
    ld12(Q, Qji_in + (size_t)e * 12);
    ld12(v, v_in + (size_t)e * 12);
    float norm = Lf;
    for (int m = 0; m < 12; ++m) norm += q[m] * Kss[m];
    float y[12], mu = 0.f;
    for (int c = 0; c < 12; ++c) {
      float an = Lf * v[c];
      for (int m = 0; m < 12; ++m) an += q[m] * Kvs[m * 12 + c];
      y[c] = Q[c] + an / norm;
      mu += y[c];
    }
    mu *= (1.f / 12.f);
    float var = 0.f;
    for (int c = 0; c < 12; ++c) { float t = y[c] - mu; var += t * t; }
    var *= (1.f / 12.f);
    float rs = rsqrtf(var + 1e-5f);
    half_t* xr = xt + lane * XS;
    cvt_nf_row(xr, nf + s * 10);
    cvt_nf_row(xr + 10, nf + d * 10);
    cvt_gt_row(xr + 20, n_ats, n_slew, n_netd, s);
    for (int j = 0; j < 12; ++j) xr[32 + j] = (half_t)((y[j] - mu) * rs * Lg[j] + Lb[j]);
    for (int j = 44; j < 64; ++j) xr[j] = (half_t)0.f;
  }
  v8f acc = mlp5_64(xt, Wt, Bs, W4t, B4);
  const int g = lane >> 4, col = lane & 15;
#pragma unroll
  for (int r = 0; r < 8; ++r) {
    int row = r + 8 * g;
    int e = e0 + row;
    if (col < 12) {
      int s = Sidx[wv][row], d = Didx[wv][row];
      float mj = acc[r] + gt_at(n_ats, n_slew, n_netd, s, col);
      atomicAdd(&msum[d * 12 + col], mj);
      atomicMaxFloat(&mmax[d * 12 + col], mj);
    } else {
      efce_out[(size_t)e * 4 + (col - 12)] = acc[r];
    }
  }
}

// per-node: mlp_a on [mean, max, nf] + final select into new_nf
__global__ __launch_bounds__(256) void mlpa_kernel(
    const float* nf, const float* n_ats, const float* n_slew, const float* n_netd,
    const unsigned char* pi_mask, const float* cnt, const float* net_deg,
    const float* net_sum, const float* msum, const float* mmax,
    MlpPtrs mp, float* new_nf, int ntiles) {
  __shared__ __align__(32) half_t Wt[4 * 64 * 64];
  __shared__ __align__(32) half_t W4t[16 * 64];
  __shared__ float Bs[4 * 64];
  __shared__ float B4[16];
  __shared__ __align__(32) half_t Xt[8][16 * XS];
  load_wt(Wt + 0 * 4096, mp.W0, 34, 64, 64, 64);
  load_wt(Wt + 1 * 4096, mp.W1, 64, 64, 64, 64);
  load_wt(Wt + 2 * 4096, mp.W2, 64, 64, 64, 64);
  load_wt(Wt + 3 * 4096, mp.W3, 64, 64, 64, 64);
  load_wt(W4t, mp.W4, 64, 12, 64, 16);
  load_bias(Bs + 0, mp.b0, 64, 64); load_bias(Bs + 64, mp.b1, 64, 64);
  load_bias(Bs + 128, mp.b2, 64, 64); load_bias(Bs + 192, mp.b3, 64, 64);
  load_bias(B4, mp.b4, 12, 16);
  __syncthreads();
  const int wv = threadIdx.x >> 5;
  const int tile = blockIdx.x * 8 + wv;
  if (tile >= ntiles) return;
  const int n0 = tile * 16;
  const int lane = threadIdx.x & 31;
  half_t* xt = Xt[wv];
  if (lane < 16) {
    int n = n0 + lane;
    float c = cnt[n];
    float denom = 1.f / fmaxf(c, 1.f);
    bool has = c > 0.f;
    float ms[12], mx[12];
    ld12(ms, msum + (size_t)n * 12);
    ld12(mx, mmax + (size_t)n * 12);
    half_t* xr = xt + lane * XS;
    for (int j = 0; j < 12; ++j) {
      xr[j] = (half_t)(ms[j] * denom);
      xr[12 + j] = (half_t)(has ? mx[j] : 0.f);
    }
    cvt_nf_row(xr + 24, nf + n * 10);
    for (int j = 34; j < 64; ++j) xr[j] = (half_t)0.f;
  }
  v8f acc = mlp5_64(xt, Wt, Bs, W4t, B4);
  const int g = lane >> 4, col = lane & 15;
  if (col < 12) {
#pragma unroll
    for (int r = 0; r < 8; ++r) {
      int n = n0 + r + 8 * g;
      float val;
      if (cnt[n] > 0.f)            val = acc[r];
      else if (net_deg[n] > 0.f)   val = net_sum[n * 12 + col];
      else if (pi_mask[n])         val = gt_at(n_ats, n_slew, n_netd, n, col);
      else                         val = 0.f;
      new_nf[(size_t)n * 12 + col] = val;
    }
  }
}

// ---------------- host ----------------
static inline MlpPtrs mk_mlp(void* const* d_in, int base) {
  MlpPtrs m;
  m.W0 = (const float*)d_in[base + 0]; m.b0 = (const float*)d_in[base + 1];
  m.W1 = (const float*)d_in[base + 2]; m.b1 = (const float*)d_in[base + 3];
  m.W2 = (const float*)d_in[base + 4]; m.b2 = (const float*)d_in[base + 5];
  m.W3 = (const float*)d_in[base + 6]; m.b3 = (const float*)d_in[base + 7];
  m.W4 = (const float*)d_in[base + 8]; m.b4 = (const float*)d_in[base + 9];
  return m;
}

extern "C" void kernel_launch(void* const* d_in, const int* in_sizes, int n_in,
                              void* d_out, int out_size, void* d_ws, size_t ws_size,
                              hipStream_t stream) {
  (void)in_sizes; (void)n_in; (void)out_size; (void)ws_size;
  const float* nf      = (const float*)d_in[0];
  const float* n_ats   = (const float*)d_in[1];
  const float* n_slew  = (const float*)d_in[2];
  const float* n_netd  = (const float*)d_in[3];
  const float* ef      = (const float*)d_in[4];
  const int* net_src   = (const int*)d_in[5];
  const int* net_dst   = (const int*)d_in[6];
  const int* cell_src  = (const int*)d_in[7];
  const int* cell_dst  = (const int*)d_in[8];
  const unsigned char* pi_mask = (const unsigned char*)d_in[9];
  // params pytree-flattened (sorted keys): Wk,Wq,Wv,bk,bq,bv,ln_b,ln_g,mlp_a,mlp_m,mlp_q,netprop
  const float* Wk   = (const float*)d_in[10];
  const float* Wq   = (const float*)d_in[11];
  const float* Wv   = (const float*)d_in[12];
  const float* bk   = (const float*)d_in[13];
  const float* bq   = (const float*)d_in[14];
  const float* bv   = (const float*)d_in[15];
  const float* ln_b = (const float*)d_in[16];
  const float* ln_g = (const float*)d_in[17];
  MlpPtrs mlp_a  = mk_mlp(d_in, 18);
  MlpPtrs mlp_m  = mk_mlp(d_in, 28);
  MlpPtrs mlp_q  = mk_mlp(d_in, 38);
  MlpPtrs netprop = mk_mlp(d_in, 48);

  float* ws = (float*)d_ws;
  float* out = (float*)d_out;             // [150000*12 new_nf | 200000*4 efce]

  const int net_tiles  = E_NETC / 16;     // 25000
  const int cell_tiles = E_CELLC / 16;    // 12500
  const int node_tiles = N_NODES / 16;    // 9375

  init_ws_kernel<<<2048, 256, 0, stream>>>(ws);
  degrees_kernel<<<(E_NETC + 255) / 256, 256, 0, stream>>>(
      net_dst, ws + OFF_NETDEG, cell_dst, ws + OFF_CNT);
  netprop_kernel<<<(net_tiles + 7) / 8, 256, 0, stream>>>(
      nf, n_ats, n_slew, n_netd, net_src, net_dst, netprop, ws + OFF_NETSUM, net_tiles);
  mlpq_kernel<<<(cell_tiles + 7) / 8, 256, 0, stream>>>(
      nf, n_ats, n_slew, n_netd, cell_src, cell_dst, mlp_q, Wq, bq,
      ws + OFF_Q, ws + OFF_QJI, ws + OFF_SUMSQ + 0, cell_tiles);
  kv_kernel<<<(cell_tiles + 7) / 8, 256, 0, stream>>>(
      ef, Wk, bk, Wv, bv, ws + OFF_V, ws + OFF_SUMSQ + 1, ws + OFF_KSSUM, ws + OFF_KVS,
      cell_tiles);
  finalize_attn<<<1, 256, 0, stream>>>(
      ws + OFF_SUMSQ, ws + OFF_KVS, ws + OFF_KSSUM, ws + OFF_KVSS, ws + OFF_KSS,
      ws + OFF_SCAL);
  mlpm_kernel<<<(cell_tiles + 7) / 8, 256, 0, stream>>>(
      nf, n_ats, n_slew, n_netd, cell_src, cell_dst,
      ws + OFF_Q, ws + OFF_QJI, ws + OFF_V, ws + OFF_KVSS, ws + OFF_KSS, ws + OFF_SCAL,
      ln_g, ln_b, mlp_m, ws + OFF_MSUM, ws + OFF_MMAX, out + (size_t)N_NODES * 12,
      cell_tiles);
  mlpa_kernel<<<(node_tiles + 7) / 8, 256, 0, stream>>>(
      nf, n_ats, n_slew, n_netd, pi_mask, ws + OFF_CNT, ws + OFF_NETDEG,
      ws + OFF_NETSUM, ws + OFF_MSUM, ws + OFF_MMAX, mlp_a, out, node_tiles);
}
